// VisionAttention_21689584845629
// MI455X (gfx1250) — compile-verified, matrix operand order checked by reading, not run on hardware
//
#include <hip/hip_runtime.h>

#define SEQ    3072
#define HIDDEN 1280
#define NHEADS 16
#define HDIM   80
#define DPAD   96
#define QKVN   3840
#define HSZ    294912   // per-head elems (SEQ*DPAD) for Qp/Kp/Vp

typedef __attribute__((ext_vector_type(16))) __bf16 v16bf;
typedef __attribute__((ext_vector_type(8)))  float  v8f;

__device__ __forceinline__ __bf16 f2bf(float f) {
  unsigned int u = __float_as_uint(f);
  unsigned int r = (u + 0x7FFFu + ((u >> 16) & 1u)) >> 16;
  unsigned short h = (unsigned short)r;
  return __builtin_bit_cast(__bf16, h);
}
__device__ __forceinline__ float bf2f(__bf16 b) {
  unsigned short h = __builtin_bit_cast(unsigned short, b);
  return __uint_as_float(((unsigned int)h) << 16);
}
// Element (m, k) of a 16-row x K fragment-ordered tile, layout [kchunk][lane 32][e 16].
// Inverse of the 16-bit WMMA A/B VGPR striping (ISA 7.12.2).
__device__ __forceinline__ int fragIdx(int m, int k) {
  int kk = k & 31;
  int g  = kk >> 3;
  return ((k >> 5) << 9) + ((m + ((g & 1) << 4)) << 4) + ((kk & 7) | ((g >> 1) << 3));
}
__device__ __forceinline__ unsigned long long pack4(float a, float b, float c, float d) {
  return (unsigned long long)__builtin_bit_cast(unsigned short, f2bf(a))
       | ((unsigned long long)__builtin_bit_cast(unsigned short, f2bf(b)) << 16)
       | ((unsigned long long)__builtin_bit_cast(unsigned short, f2bf(c)) << 32)
       | ((unsigned long long)__builtin_bit_cast(unsigned short, f2bf(d)) << 48);
}

// ---------------------------------------------------------------- zero init
__global__ __launch_bounds__(256) void zero_ws_kernel(unsigned long long* p, long long n) {
  long long i = (long long)blockIdx.x * blockDim.x + threadIdx.x;
  if (i < n) p[i] = 0ull;
}

// ---------------- QKV GEMM + bias -> fragment-ordered bf16 Q/K/V (head-dim padded to 96)
__global__ __launch_bounds__(256)
void qkv_gemm_pack(const float* __restrict__ A, const float* __restrict__ W,
                   const float* __restrict__ bias,
                   __bf16* __restrict__ Qp, __bf16* __restrict__ Kp, __bf16* __restrict__ Vp) {
  __shared__ v16bf AsV[256];  // 8 row-tiles x [lane][e]
  __shared__ v16bf BsV[128];  // 4 col-tiles x [lane][e]
  __bf16* As = (__bf16*)AsV;
  __bf16* Bs = (__bf16*)BsV;
  const int tid = threadIdx.x, lane = tid & 31, wave = tid >> 5;
  const int ln = lane & 15, half = lane >> 4;
  const int m0 = blockIdx.x * 128, n0 = blockIdx.y * 64;
  v8f acc[4] = {};

  for (int k0 = 0; k0 < HIDDEN; k0 += 32) {
    if (k0 + 32 < HIDDEN)
      __builtin_prefetch(&W[(size_t)(k0 + 32 + (tid >> 6)) * QKVN + n0], 0, 1);
    // A tile 128x32 f32 -> bf16, fragment order, 4x b64 stores per thread
#pragma unroll
    for (int t = 0; t < 4; ++t) {
      int gi = tid + t * 256;            // 1024 groups of 4 consecutive k
      int r = gi >> 3, c4 = (gi & 7) << 2;
      const float4 v = *(const float4*)&A[(size_t)(m0 + r) * HIDDEN + k0 + c4];
      *(unsigned long long*)&As[((r >> 4) << 9) + fragIdx(r & 15, c4)] =
          pack4(v.x, v.y, v.z, v.w);
    }
    // B tile 32x64: 4 strided (coalesced across nc) loads -> one b64 store
#pragma unroll
    for (int t = 0; t < 2; ++t) {
      int gi = tid + t * 256;            // 512 groups of 4 consecutive k-rows
      int nc = gi & 63, kr0 = (gi >> 6) << 2;
      const float* wp = &W[(size_t)(k0 + kr0) * QKVN + n0 + nc];
      *(unsigned long long*)&Bs[((nc >> 4) << 9) + fragIdx(nc & 15, kr0)] =
          pack4(wp[0], wp[QKVN], wp[2 * QKVN], wp[3 * QKVN]);
    }
    __syncthreads();
    v16bf a = *(const v16bf*)&As[(wave << 9) + (lane << 4)];
#pragma unroll
    for (int nt = 0; nt < 4; ++nt) {
      v16bf b = *(const v16bf*)&Bs[(nt << 9) + (lane << 4)];
      acc[nt] = __builtin_amdgcn_wmma_f32_16x16x32_bf16(false, a, false, b,
                                                        (short)0, acc[nt], false, false);
    }
    __syncthreads();
  }
#pragma unroll
  for (int nt = 0; nt < 4; ++nt) {
    int col = n0 + nt * 16 + ln;
    float bv = bias[col];
    int sel = col / HIDDEN;
    int rem = col - sel * HIDDEN;
    int hd = rem / HDIM;
    int d = rem - hd * HDIM;
    __bf16* dst = (sel == 0) ? Qp : ((sel == 1) ? Kp : Vp);
    const size_t hb = (size_t)hd * HSZ;
#pragma unroll
    for (int r = 0; r < 8; ++r) {
      int row = m0 + wave * 16 + r + half * 8;
      size_t off = (sel < 2)
          ? hb + (size_t)(row >> 4) * 1536 + fragIdx(row & 15, d)                 // Q/K: A-frag order
          : hb + (size_t)(row >> 6) * 6144 + (size_t)(d >> 4) * 1024 + fragIdx(d & 15, row & 63); // V: B-frag order
      dst[off] = f2bf(acc[nt][r] + bv);
    }
  }
}

// ---------------------------------------------------------------- 2D RoPE on Q,K in place
__global__ __launch_bounds__(256)
void rope_kernel(const int* __restrict__ pos, __bf16* __restrict__ Qp, __bf16* __restrict__ Kp) {
  int id = blockIdx.x * blockDim.x + threadIdx.x;
  if (id >= SEQ * NHEADS * 40) return;
  int j = id % 40;
  int h = (id / 40) % NHEADS;
  int s = id / (40 * NHEADS);
  float p = (float)((j < 20) ? pos[s * 2 + 0] : pos[s * 2 + 1]);
  int i = j % 20;
  float invf = __expf(-(float)i * 0.46051701859880917f);  // 10000^(-i/20)
  float f = p * invf;
  float c = __cosf(f), sn = __sinf(f);
  size_t base = (size_t)h * HSZ + (size_t)(s >> 4) * 1536;
  int m = s & 15;
  size_t i1 = base + fragIdx(m, j);
  size_t i2 = base + fragIdx(m, j + 40);
  const float scaling = 0.11180339887498948f;             // 80^-0.5 folded into Q
  {
    float x1 = bf2f(Qp[i1]), x2 = bf2f(Qp[i2]);
    Qp[i1] = f2bf((x1 * c - x2 * sn) * scaling);
    Qp[i2] = f2bf((x2 * c + x1 * sn) * scaling);
  }
  {
    float x1 = bf2f(Kp[i1]), x2 = bf2f(Kp[i2]);
    Kp[i1] = f2bf(x1 * c - x2 * sn);
    Kp[i2] = f2bf(x2 * c + x1 * sn);
  }
}

// ---------------- flash attention: async double-buffered K/V tiles, all-WMMA core
__global__ __launch_bounds__(256)
void flash_attn(const __bf16* __restrict__ Qp, const __bf16* __restrict__ Kp,
                const __bf16* __restrict__ Vp, __bf16* __restrict__ O) {
  __shared__ v16bf KsV[2][384];   // 64 keys x 96 dims, fragment order (12 KB each)
  __shared__ v16bf VsV[2][384];
  __shared__ v16bf PsV[8][64];    // per-wave P staging, A-frag order (2 KB each)
  const int tid = threadIdx.x, lane = tid & 31, wave = tid >> 5;
  const int ln = lane & 15, half = lane >> 4;
  const int head = blockIdx.y;
  const int q0 = blockIdx.x * 128;
  const size_t hbase = (size_t)head * HSZ;
  const __bf16* Kh = Kp + hbase;
  const __bf16* Vh = Vp + hbase;

  auto issue = [&](int buf, int kbi) {
    const __bf16* kg = Kh + (size_t)kbi * 6144;   // identity copy: global layout == LDS layout
    const __bf16* vg = Vh + (size_t)kbi * 6144;
    char* kl = (char*)&KsV[buf][0];
    char* vl = (char*)&VsV[buf][0];
    for (int i = tid; i < 768; i += 256) {        // 768 x 16B per tile
      asm volatile("global_load_async_to_lds_b128 %0, %1, off"
                   :: "v"((unsigned)(size_t)(kl + i * 16)),
                      "v"((unsigned long long)(size_t)(kg + i * 8))
                   : "memory");
      asm volatile("global_load_async_to_lds_b128 %0, %1, off"
                   :: "v"((unsigned)(size_t)(vl + i * 16)),
                      "v"((unsigned long long)(size_t)(vg + i * 8))
                   : "memory");
    }
  };

  issue(0, 0);

  // Q fragments straight from global (fragment-ordered, L2-resident)
  v16bf qf[3];
  const __bf16* Qt = Qp + hbase + (size_t)(q0 / 16 + wave) * 1536;
#pragma unroll
  for (int kc = 0; kc < 3; ++kc)
    qf[kc] = *(const v16bf*)&Qt[(kc << 9) + (lane << 4)];

  float m_i[8], l_i[8];
#pragma unroll
  for (int r = 0; r < 8; ++r) { m_i[r] = -1e30f; l_i[r] = 0.f; }
  v8f oacc[5] = {};

  for (int kbi = 0; kbi < SEQ / 64; ++kbi) {
    const int cur = kbi & 1;
    asm volatile("s_wait_asynccnt 0x0" ::: "memory");
    __syncthreads();
    if (kbi + 1 < SEQ / 64) issue(cur ^ 1, kbi + 1);  // overlap next DMA with compute
    const __bf16* Ksb = (const __bf16*)&KsV[cur][0];
    const __bf16* Vsb = (const __bf16*)&VsV[cur][0];
    __bf16* Psb = (__bf16*)&PsV[wave][0];

    v8f s[4] = {};
#pragma unroll
    for (int kc = 0; kc < 3; ++kc) {
#pragma unroll
      for (int nt = 0; nt < 4; ++nt) {
        v16bf kf = *(const v16bf*)&Ksb[nt * 1536 + (kc << 9) + (lane << 4)];
        s[nt] = __builtin_amdgcn_wmma_f32_16x16x32_bf16(false, qf[kc], false, kf,
                                                        (short)0, s[nt], false, false);
      }
    }
    // online softmax; row (r + 8*half) lives on a 16-lane group
#pragma unroll
    for (int r = 0; r < 8; ++r) {
      float rmax = s[0][r];
#pragma unroll
      for (int nt = 1; nt < 4; ++nt) rmax = fmaxf(rmax, s[nt][r]);
#pragma unroll
      for (int off = 1; off < 16; off <<= 1)
        rmax = fmaxf(rmax, __shfl_xor(rmax, off, 32));
      float mn = fmaxf(m_i[r], rmax);
      float alpha = __expf(m_i[r] - mn);
      m_i[r] = mn;
      float rsum = 0.f;
#pragma unroll
      for (int nt = 0; nt < 4; ++nt) {
        float pv = __expf(s[nt][r] - mn);
        rsum += pv;
        Psb[fragIdx(r + half * 8, nt * 16 + ln)] = f2bf(pv);
      }
#pragma unroll
      for (int off = 1; off < 16; off <<= 1)
        rsum += __shfl_xor(rsum, off, 32);
      l_i[r] = l_i[r] * alpha + rsum;
#pragma unroll
      for (int t = 0; t < 5; ++t) oacc[t][r] = oacc[t][r] * alpha;
    }
    v16bf pf[2];
#pragma unroll
    for (int kc = 0; kc < 2; ++kc)
      pf[kc] = *(const v16bf*)&Psb[(kc << 9) + (lane << 4)];
#pragma unroll
    for (int nt = 0; nt < 5; ++nt) {
#pragma unroll
      for (int kc = 0; kc < 2; ++kc) {
        v16bf vf = *(const v16bf*)&Vsb[nt * 1024 + (kc << 9) + (lane << 4)];
        oacc[nt] = __builtin_amdgcn_wmma_f32_16x16x32_bf16(false, pf[kc], false, vf,
                                                           (short)0, oacc[nt], false, false);
      }
    }
  }
  // write attn output in O-proj A-fragment order: [row-tile 192][kc 40][lane][e]
  const size_t mtb = (size_t)(q0 / 16 + wave) * 20480;
#pragma unroll
  for (int nt = 0; nt < 5; ++nt) {
#pragma unroll
    for (int r = 0; r < 8; ++r) {
      int m = r + half * 8;
      int col = head * HDIM + nt * 16 + ln;    // < 1280
      O[mtb + fragIdx(m, col)] = f2bf(oacc[nt][r] / l_i[r]);
    }
  }
}

// ---------------------------------------------------------------- output projection GEMM
__global__ __launch_bounds__(256)
void out_gemm(const __bf16* __restrict__ Aat, const float* __restrict__ W,
              const float* __restrict__ bias, float* __restrict__ C) {
  __shared__ v16bf BsV[128];
  __bf16* Bs = (__bf16*)BsV;
  const int tid = threadIdx.x, lane = tid & 31, wave = tid >> 5;
  const int ln = lane & 15, half = lane >> 4;
  const int m0 = blockIdx.x * 128, n0 = blockIdx.y * 64;
  const size_t abase = (size_t)(blockIdx.x * 8 + wave) * 20480;  // A frags direct from global
  v8f acc[4] = {};

  for (int k0 = 0; k0 < HIDDEN; k0 += 32) {
    if (k0 + 32 < HIDDEN)
      __builtin_prefetch(&W[(size_t)(k0 + 32 + (tid >> 6)) * HIDDEN + n0], 0, 1);
#pragma unroll
    for (int t = 0; t < 2; ++t) {
      int gi = tid + t * 256;
      int nc = gi & 63, kr0 = (gi >> 6) << 2;
      const float* wp = &W[(size_t)(k0 + kr0) * HIDDEN + n0 + nc];
      *(unsigned long long*)&Bs[((nc >> 4) << 9) + fragIdx(nc & 15, kr0)] =
          pack4(wp[0], wp[HIDDEN], wp[2 * HIDDEN], wp[3 * HIDDEN]);
    }
    __syncthreads();
    v16bf a = *(const v16bf*)&Aat[abase + ((size_t)(k0 >> 5) << 9) + (lane << 4)];
#pragma unroll
    for (int nt = 0; nt < 4; ++nt) {
      v16bf b = *(const v16bf*)&Bs[(nt << 9) + (lane << 4)];
      acc[nt] = __builtin_amdgcn_wmma_f32_16x16x32_bf16(false, a, false, b,
                                                        (short)0, acc[nt], false, false);
    }
    __syncthreads();
  }
#pragma unroll
  for (int nt = 0; nt < 4; ++nt) {
    int col = n0 + nt * 16 + ln;
    float bv = bias[col];
#pragma unroll
    for (int r = 0; r < 8; ++r) {
      int row = m0 + wave * 16 + r + half * 8;
      C[(size_t)row * HIDDEN + col] = acc[nt][r] + bv;
    }
  }
}

extern "C" void kernel_launch(void* const* d_in, const int* in_sizes, int n_in,
                              void* d_out, int out_size, void* d_ws, size_t ws_size,
                              hipStream_t stream) {
  const float* hidden = (const float*)d_in[0];
  const int*   pos    = (const int*)d_in[1];
  const float* qkv_w  = (const float*)d_in[2];
  const float* qkv_b  = (const float*)d_in[3];
  const float* o_w    = (const float*)d_in[4];
  const float* o_b    = (const float*)d_in[5];
  float* out = (float*)d_out;

  const size_t perBuf = (size_t)NHEADS * HSZ;           // 4,718,592 bf16 elems
  __bf16* Qp = (__bf16*)d_ws;
  __bf16* Kp = Qp + perBuf;
  __bf16* Vp = Kp + perBuf;
  __bf16* attnout = Vp + perBuf;                        // 192*40*512 bf16 elems

  long long qwords = (long long)(3 * perBuf * sizeof(__bf16) / 8);
  zero_ws_kernel<<<(unsigned)((qwords + 255) / 256), 256, 0, stream>>>(
      (unsigned long long*)d_ws, qwords);

  dim3 g1(SEQ / 128, QKVN / 64);
  qkv_gemm_pack<<<g1, 256, 0, stream>>>(hidden, qkv_w, qkv_b, Qp, Kp, Vp);

  int nrope = SEQ * NHEADS * 40;
  rope_kernel<<<(nrope + 255) / 256, 256, 0, stream>>>(pos, Qp, Kp);

  dim3 g3(SEQ / 128, NHEADS);
  flash_attn<<<g3, 256, 0, stream>>>(Qp, Kp, Vp, attnout);

  dim3 g4(SEQ / 128, HIDDEN / 64);
  out_gemm<<<g4, 256, 0, stream>>>(attnout, o_w, o_b, out);
}
